// WanSelfAttention_37993280701157
// MI455X (gfx1250) — matrix operand order repaired and checked by reading, not compiled, not run on hardware
//
#include <hip/hip_runtime.h>
#include <hip/hip_bf16.h>

// ---------------- problem constants ----------------
#define B_    2
#define L_    2048
#define DIM_  1536
#define H_    12
#define HD_   128
#define MTOK  (B_ * L_)          // 4096 tokens
#define EPS_  1e-6f
#define SCALE_ 0.08838834764831845f   // 1/sqrt(128), folded into q at RoPE time

// ---------------- vector types ----------------
typedef __bf16 bf8v  __attribute__((ext_vector_type(8)));
typedef __bf16 bf16v __attribute__((ext_vector_type(16)));
typedef float  f8v   __attribute__((ext_vector_type(8)));

#define WMMA_BF16(a, b, c) \
  __builtin_amdgcn_wmma_f32_16x16x32_bf16(false, (a), false, (b), (short)0, (c), false, false)

// Build a 16-element bf16 A-fragment from two contiguous 8-element chunks.
static __device__ __forceinline__ bf16v ld2(const __bf16* p0, const __bf16* p1) {
  bf8v a = *(const bf8v*)p0;
  bf8v b = *(const bf8v*)p1;
  bf16v r;
#pragma unroll
  for (int i = 0; i < 8; ++i) { r[i] = a[i]; r[i + 8] = b[i]; }
  return r;
}

// ---------------- f32 -> bf16 convert ----------------
__global__ void wan_cvt_bf16(const float* __restrict__ in, __bf16* __restrict__ out, int n) {
  int i = blockIdx.x * blockDim.x + threadIdx.x;
  if (i < n) out[i] = (__bf16)in[i];
}

// ---------------- GEMM: C[M,N] = A[M,K](bf16) * W[N,K]^T(bf16) + bias, fp32 out ------------
// Wave computes a 16x64 tile. Block = 8 waves = 16 x 512. grid = (M/16, N/512).
// Register double-buffered over the K loop so loads overlap the WMMAs.
__global__ __launch_bounds__(256)
void wan_gemm_bf16(const __bf16* __restrict__ A, const __bf16* __restrict__ W,
                   const float* __restrict__ bias, float* __restrict__ C,
                   int M, int N, int K) {
  const int lane = threadIdx.x & 31;
  const int wv   = threadIdx.x >> 5;
  const int half = lane >> 4;
  const int lm   = lane & 15;
  const int m0   = blockIdx.x * 16;
  const int n0   = blockIdx.y * 512 + wv * 64;

  f8v acc[4];
  const f8v z = {0.f, 0.f, 0.f, 0.f, 0.f, 0.f, 0.f, 0.f};
#pragma unroll
  for (int s = 0; s < 4; ++s) acc[s] = z;

  const __bf16* arow = A + (size_t)(m0 + lm) * K + half * 16;  // chunks at +0 and +16B
  const __bf16* wrow[4];
#pragma unroll
  for (int s = 0; s < 4; ++s)
    wrow[s] = W + (size_t)(n0 + s * 16 + lm) * K + half * 16;  // one contiguous 32B chunk

  // prologue: fragments for k0 = 0
  bf16v af = ld2(arow, arow + 8);
  bf16v bfv[4];
#pragma unroll
  for (int s = 0; s < 4; ++s) bfv[s] = *(const bf16v*)(wrow[s]);

  for (int k0 = 0; k0 < K; k0 += 32) {
    int kn = k0 + 32;
    if (kn >= K) kn = 0;  // harmless dummy refill on last iteration (uniform)
    // issue next-step loads first ...
    bf16v afn = ld2(arow + kn, arow + kn + 8);
    bf16v bfn[4];
#pragma unroll
    for (int s = 0; s < 4; ++s) bfn[s] = *(const bf16v*)(wrow[s] + kn);
    __builtin_prefetch(arow + k0 + 256, 0, 1);
    // ... then compute on current fragments while they are in flight
#pragma unroll
    for (int s = 0; s < 4; ++s) acc[s] = WMMA_BF16(af, bfv[s], acc[s]);
    af = afn;
#pragma unroll
    for (int s = 0; s < 4; ++s) bfv[s] = bfn[s];
  }

#pragma unroll
  for (int s = 0; s < 4; ++s) {
    const int col = n0 + s * 16 + lm;
    const float bs = bias ? bias[col] : 0.f;
#pragma unroll
    for (int r = 0; r < 8; ++r) {
      // C layout: VGPR r -> row r + 8*half, col = lane&15 within tile
      C[(size_t)(m0 + r + half * 8) * N + col] = acc[s][r] + bs;
    }
  }
}

// ------------- fused RMSNorm (over DIM) + RoPE epilogue; also lays out q/k/v -------------
// One block (256 thr) per token. q,k -> bf16 [B,H,L,D]; v -> bf16 transposed [B,H,D,L].
// The attention softmax scale 1/sqrt(d) is folded into q here (RoPE is linear).
__global__ __launch_bounds__(256)
void wan_norm_rope(const float* __restrict__ qraw, const float* __restrict__ kraw,
                   const float* __restrict__ vraw, const float* __restrict__ gq,
                   const float* __restrict__ gk, const float* __restrict__ freqs,
                   __bf16* __restrict__ qb, __bf16* __restrict__ kb,
                   __bf16* __restrict__ vtb) {
  const int t   = blockIdx.x;            // token id
  const int b   = t / L_;
  const int s   = t % L_;
  const int tid = threadIdx.x;

  __shared__ float redq[256];
  __shared__ float redk[256];

  const float* qr = qraw + (size_t)t * DIM_;
  const float* kr = kraw + (size_t)t * DIM_;
  const float* vr = vraw + (size_t)t * DIM_;

  float sq = 0.f, sk = 0.f;
  for (int i = tid; i < DIM_; i += 256) {
    float a = qr[i]; sq += a * a;
    float c = kr[i]; sk += c * c;
  }
  redq[tid] = sq; redk[tid] = sk;
  __syncthreads();
  for (int o = 128; o > 0; o >>= 1) {
    if (tid < o) { redq[tid] += redq[tid + o]; redk[tid] += redk[tid + o]; }
    __syncthreads();
  }
  const float rq = rsqrtf(redq[0] / (float)DIM_ + EPS_) * SCALE_;  // fold softmax scale
  const float rk = rsqrtf(redk[0] / (float)DIM_ + EPS_);

  const float* fz = freqs + (size_t)s * (2 * HD_);

  // RoPE over interleaved pairs
  for (int p = tid; p < DIM_ / 2; p += 256) {
    const int i0 = 2 * p, i1 = 2 * p + 1;
    const int h = i0 >> 7, d0 = i0 & 127, d1 = d0 + 1;
    const float c0 = fz[d0], c1 = fz[d1];
    const float s0 = fz[HD_ + d0], s1 = fz[HD_ + d1];

    float qy0 = qr[i0] * rq * gq[i0];
    float qy1 = qr[i1] * rq * gq[i1];
    float ky0 = kr[i0] * rk * gk[i0];
    float ky1 = kr[i1] * rk * gk[i1];

    const size_t base = (((size_t)b * H_ + h) * L_ + s) * HD_;
    qb[base + d0] = (__bf16)(qy0 * c0 - qy1 * s0);
    qb[base + d1] = (__bf16)(qy1 * c1 + qy0 * s1);
    kb[base + d0] = (__bf16)(ky0 * c0 - ky1 * s0);
    kb[base + d1] = (__bf16)(ky1 * c1 + ky0 * s1);
  }

  // v: transpose into [B,H,D,L]
  for (int i = tid; i < DIM_; i += 256) {
    const int h = i >> 7, d = i & 127;
    vtb[(((size_t)b * H_ + h) * HD_ + d) * L_ + s] = (__bf16)vr[i];
  }
}

// ---------------- flash attention ----------------
// grid (L/128, H, B), block 256 (8 waves). Wave w handles 16 q-rows.
// No block barriers: each wave owns its LDS slice; intra-wave DS ordering is
// enforced with s_wait_dscnt (CDNA5 split counter) + compiler fences.
__global__ __launch_bounds__(256)
void wan_attn(const __bf16* __restrict__ qb, const __bf16* __restrict__ kb,
              const __bf16* __restrict__ vtb, const int* __restrict__ seq_lens,
              __bf16* __restrict__ attnb) {
  const int lane = threadIdx.x & 31;
  const int w    = threadIdx.x >> 5;
  const int half = lane >> 4;
  const int lm   = lane & 15;
  const int b    = blockIdx.z;
  const int h    = blockIdx.y;
  const int q0   = blockIdx.x * 128 + w * 16;

  const int slen   = seq_lens[b];
  const int ktiles = (slen + 31) >> 5;

  __shared__ __align__(64) __bf16 lds_p[8][16][32];

  const __bf16* qh   = qb  + ((size_t)b * H_ + h) * L_ * HD_;
  const __bf16* krow = kb  + ((size_t)b * H_ + h) * L_ * HD_ + (size_t)lm * HD_ + half * 16;
  const __bf16* vrow = vtb + ((size_t)b * H_ + h) * HD_ * L_ + (size_t)lm * L_ + half * 16;

  // Preload Q fragments: 4 k-steps covering D=128 (already scaled by 1/sqrt(d))
  bf16v qf[4];
#pragma unroll
  for (int c = 0; c < 4; ++c) {
    const __bf16* qp = qh + (size_t)(q0 + lm) * HD_ + c * 32 + half * 16;
    qf[c] = ld2(qp, qp + 8);
  }

  const f8v z = {0.f, 0.f, 0.f, 0.f, 0.f, 0.f, 0.f, 0.f};
  f8v of[8];
#pragma unroll
  for (int nt = 0; nt < 8; ++nt) of[nt] = z;
  float m8[8], l8[8];
#pragma unroll
  for (int r = 0; r < 8; ++r) { m8[r] = -3.0e38f; l8[r] = 0.f; }

  for (int kt = 0; kt < ktiles; ++kt) {
    const int kbase = kt * 32;

    // ---- batch all K-fragment loads, then the 8 S WMMAs ----
    const __bf16* kp = krow + (size_t)kbase * HD_;
    bf16v kf0[4], kf1[4];
#pragma unroll
    for (int c = 0; c < 4; ++c) {
      kf0[c] = *(const bf16v*)(kp + c * 32);
      kf1[c] = *(const bf16v*)(kp + 16 * HD_ + c * 32);
    }
    f8v s0 = z, s1 = z;
#pragma unroll
    for (int c = 0; c < 4; ++c) {
      s0 = WMMA_BF16(qf[c], kf0[c], s0);
      s1 = WMMA_BF16(qf[c], kf1[c], s1);
    }

    // ---- issue V-fragment loads early: latency hides under softmax VALU ----
    bf16v vf[8];
#pragma unroll
    for (int nt = 0; nt < 8; ++nt)
      vf[nt] = *(const bf16v*)(vrow + (size_t)(nt * 16) * L_ + kbase);

    const bool ok0 = (kbase + lm) < slen;
    const bool ok1 = (kbase + 16 + lm) < slen;

    float al[8];
#pragma unroll
    for (int r = 0; r < 8; ++r) {
      float a = ok0 ? s0[r] : -3.0e38f;
      float c = ok1 ? s1[r] : -3.0e38f;
      float tmax = fmaxf(a, c);
#pragma unroll
      for (int o = 1; o < 16; o <<= 1) tmax = fmaxf(tmax, __shfl_xor(tmax, o, 32));
      const float mn = fmaxf(m8[r], tmax);
      const float alpha = __expf(m8[r] - mn);
      const float p0 = __expf(a - mn);
      const float p1 = __expf(c - mn);
      float rs = p0 + p1;
#pragma unroll
      for (int o = 1; o < 16; o <<= 1) rs += __shfl_xor(rs, o, 32);
      l8[r] = l8[r] * alpha + rs;
      m8[r] = mn;
      al[r] = alpha;
      const int row = r + half * 8;
      lds_p[w][row][lm]      = (__bf16)p0;
      lds_p[w][row][16 + lm] = (__bf16)p1;
    }
#pragma unroll
    for (int nt = 0; nt < 8; ++nt)
#pragma unroll
      for (int r = 0; r < 8; ++r) of[nt][r] *= al[r];

    // Intra-wave DS store->load ordering: drain DScnt, fence the compiler.
    asm volatile("s_wait_dscnt 0x0" ::: "memory");

    // Re-layout P (C-fragment) -> bf16 A-fragment via this wave's LDS slice
    bf16v pf;
#pragma unroll
    for (int e = 0; e < 8; ++e) {
      pf[e]     = lds_p[w][lm][half * 16 + e];
      pf[8 + e] = lds_p[w][lm][half * 16 + 8 + e];
    }
    asm volatile("" ::: "memory");  // keep next iter's stores after these reads

    // O += P * V
#pragma unroll
    for (int nt = 0; nt < 8; ++nt) of[nt] = WMMA_BF16(pf, vf[nt], of[nt]);
  }

  float inv[8];
#pragma unroll
  for (int r = 0; r < 8; ++r) inv[r] = 1.0f / l8[r];

#pragma unroll
  for (int nt = 0; nt < 8; ++nt) {
#pragma unroll
    for (int r = 0; r < 8; ++r) {
      const int row = q0 + r + half * 8;
      attnb[((size_t)b * L_ + row) * DIM_ + h * HD_ + nt * 16 + lm] =
          (__bf16)(of[nt][r] * inv[r]);
    }
  }
}

// ---------------- host launch ----------------
extern "C" void kernel_launch(void* const* d_in, const int* in_sizes, int n_in,
                              void* d_out, int out_size, void* d_ws, size_t ws_size,
                              hipStream_t stream) {
  (void)in_sizes; (void)n_in; (void)out_size; (void)ws_size;
  const float* x  = (const float*)d_in[0];
  const float* wq = (const float*)d_in[1];
  const float* bq = (const float*)d_in[2];
  const float* wk = (const float*)d_in[3];
  const float* bk = (const float*)d_in[4];
  const float* wv = (const float*)d_in[5];
  const float* bv = (const float*)d_in[6];
  const float* wo = (const float*)d_in[7];
  const float* bo = (const float*)d_in[8];
  const float* gq = (const float*)d_in[9];
  const float* gk = (const float*)d_in[10];
  const float* fr = (const float*)d_in[11];
  const int* seq_lens = (const int*)d_in[12];
  // d_in[13] grid_sizes: unused (full attention window)

  char* ws = (char*)d_ws;
  size_t off = 0;
  auto take = [&](size_t bytes) -> void* {
    void* p = ws + off;
    off += (bytes + 255) & ~(size_t)255;
    return p;
  };

  const size_t NTOK = (size_t)MTOK * DIM_;
  const size_t NW   = (size_t)DIM_ * DIM_;

  __bf16* xb    = (__bf16*)take(NTOK * 2);
  __bf16* wqb   = (__bf16*)take(NW * 2);
  __bf16* wkb   = (__bf16*)take(NW * 2);
  __bf16* wvb   = (__bf16*)take(NW * 2);
  __bf16* wob   = (__bf16*)take(NW * 2);
  float*  qraw  = (float*)take(NTOK * 4);
  float*  kraw  = (float*)take(NTOK * 4);
  float*  vraw  = (float*)take(NTOK * 4);
  __bf16* qb    = (__bf16*)take(NTOK * 2);
  __bf16* kbuf  = (__bf16*)take(NTOK * 2);
  __bf16* vtb   = (__bf16*)take(NTOK * 2);
  __bf16* attnb = (__bf16*)take(NTOK * 2);

  // 1) converts
  wan_cvt_bf16<<<(int)((NTOK + 255) / 256), 256, 0, stream>>>(x, xb, (int)NTOK);
  wan_cvt_bf16<<<(int)((NW + 255) / 256), 256, 0, stream>>>(wq, wqb, (int)NW);
  wan_cvt_bf16<<<(int)((NW + 255) / 256), 256, 0, stream>>>(wk, wkb, (int)NW);
  wan_cvt_bf16<<<(int)((NW + 255) / 256), 256, 0, stream>>>(wv, wvb, (int)NW);
  wan_cvt_bf16<<<(int)((NW + 255) / 256), 256, 0, stream>>>(wo, wob, (int)NW);

  // 2) QKV projections (fp32 out for exact RMSNorm)
  dim3 gg(MTOK / 16, DIM_ / 512);
  wan_gemm_bf16<<<gg, 256, 0, stream>>>(xb, wqb, bq, qraw, MTOK, DIM_, DIM_);
  wan_gemm_bf16<<<gg, 256, 0, stream>>>(xb, wkb, bk, kraw, MTOK, DIM_, DIM_);
  wan_gemm_bf16<<<gg, 256, 0, stream>>>(xb, wvb, bv, vraw, MTOK, DIM_, DIM_);

  // 3) RMSNorm + RoPE + layout (softmax scale folded into q)
  wan_norm_rope<<<MTOK, 256, 0, stream>>>(qraw, kraw, vraw, gq, gk, fr, qb, kbuf, vtb);

  // 4) flash attention
  wan_attn<<<dim3(L_ / 128, H_, B_), 256, 0, stream>>>(qb, kbuf, vtb, seq_lens, attnb);

  // 5) output projection straight into d_out (fp32)
  wan_gemm_bf16<<<gg, 256, 0, stream>>>(attnb, wob, bo, (float*)d_out, MTOK, DIM_, DIM_);
}
